// ReverseContentGCN_35553739276401
// MI455X (gfx1250) — matrix operand order repaired and checked by reading, not compiled, not run on hardware
//
#include <hip/hip_runtime.h>
#include <math.h>

// ---------------------------------------------------------------------------
// ReverseContentGCN for MI455X (gfx1250, wave32, WMMA + TDM).
// Dominant work: 3x pointwise 512x512 1x1 convs over 224 images x 196 px
// (~23 GFLOP each) -> v_wmma_f32_16x16x32_f16 GEMMs (f16 operands, f32 acc).
// Activation B-tiles are DMA'd global->LDS by the Tensor Data Mover with
// hardware LDS padding (520-half row pitch, conflict-free ds_load_b128),
// synchronized via TENSORcnt. Everything else (pool/normalize/softmax/
// top-k/depthwise 3x3/node mixing) is bandwidth-trivial scalar kernels.
// ---------------------------------------------------------------------------

typedef _Float16 v16h __attribute__((ext_vector_type(16)));
typedef _Float16 v8h  __attribute__((ext_vector_type(8)));
typedef float    v8f  __attribute__((ext_vector_type(8)));
typedef unsigned int u32x4 __attribute__((ext_vector_type(4)));
typedef int          i32x4 __attribute__((ext_vector_type(4)));
typedef int          i32x8 __attribute__((ext_vector_type(8)));

namespace rcg {

constexpr int BB   = 4;          // batch
constexpr int HH   = 4;          // heads
constexpr int TT   = 8;          // T
constexpr int DD   = 7;          // D
constexpr int KK   = 3;          // K (top-k)
constexpr int CC   = 128;        // channels per head
constexpr int HW   = 14;         // spatial side
constexpr int SP   = HW * HW;    // 196 pixels
constexpr int SPP  = 208;        // padded to 13*16 for WMMA n-tiles
constexpr int NALL = 1 + TT*DD;  // 57 nodes
constexpr int BT   = BB * TT;    // 32
constexpr int NIMG = BT * DD;    // 224 big-stage images
constexpr int NCH  = HH * CC;    // 512 channels (= GEMM M and K)
constexpr int NTIL = SPP / 16;   // 13 n-tiles
constexpr int KST  = NCH / 32;   // 16 k-steps of 32

// ---------------------------------------------------------------------------
// Pack f32 pointwise weights [o][c] (512x512) into the CDNA5 16-bit A-matrix
// VGPR arrangement: layout [mt(32)][ks(16)][lane(32)][16 halves].
// lane<16:  M = mt*16+lane,     halves h<8 -> K = ks*32 + h, h>=8 -> K = ks*32 + 16 + (h-8)
// lane>=16: M = mt*16+lane-16,  halves h<8 -> K = ks*32 + 8 + h, h>=8 -> K = ks*32 + 24 + (h-8)
// ---------------------------------------------------------------------------
__global__ __launch_bounds__(256) void k_pack_w(const float* __restrict__ W,
                                                _Float16* __restrict__ Wp) {
  int fid  = blockIdx.x * 256 + threadIdx.x;   // 0 .. 262143
  int h    = fid & 15;
  int lane = (fid >> 4) & 31;
  int ks   = (fid >> 9) & 15;
  int mt   = fid >> 13;
  int m, kb;
  if (lane < 16) { m = mt*16 + lane;      kb = (h < 8) ? h      : 16 + (h - 8); }
  else           { m = mt*16 + lane - 16; kb = (h < 8) ? 8 + h  : 24 + (h - 8); }
  int k = ks*32 + kb;
  Wp[fid] = (_Float16)W[(size_t)m * NCH + k];
}

// ---------------------------------------------------------------------------
// WMMA pointwise GEMM: out[img][p][o] = sum_c W[o][c]*Zt[img][p][c] + bias[o]
// (+ optional residual, same layout). Zt is f16 channel-last, padded to SPP.
// Block: 256 threads = 8 waves, one 16-pixel n-tile per block.
// B-tile (16 rows x 512 halves) is loaded by one TDM descriptor into LDS
// with hardware padding of 4 DWORDs every 256 DWORDs -> 520-half row pitch.
// ---------------------------------------------------------------------------
__global__ __launch_bounds__(256) void k_pw_gemm(const _Float16* __restrict__ Wp,
                                                 const _Float16* __restrict__ Zt,
                                                 const float* __restrict__ bias,
                                                 const float* __restrict__ resid,
                                                 float* __restrict__ out) {
  __shared__ _Float16 ldsB[16 * 520];
  const int img   = blockIdx.x;
  const int pbase = blockIdx.y * 16;
  const int tid   = threadIdx.x;
  const int wave  = tid >> 5;
  const int lane  = tid & 31;
  const int pl    = lane & 15;

  // ---- TDM: DMA the 16x512 f16 tile global -> LDS (wave 0 issues) ----
  if (wave == 0) {
    unsigned long long gaddr =
        (unsigned long long)(size_t)(Zt + ((size_t)img * SPP + pbase) * NCH);
    unsigned int laddr = (unsigned int)(size_t)(&ldsB[0]);

    // D# group 0: count=1 (valid), lds_addr, global_addr[56:0], type=2.
    u32x4 g0;
    g0.x = 1u;
    g0.y = laddr;
    g0.z = (unsigned int)gaddr;                                   // addr[31:0]
    g0.w = ((unsigned int)(gaddr >> 32) & 0x01FFFFFFu) | (2u << 30); // addr[56:32] | type

    // D# group 1: data_size=2B, pad_enable, pad every 256 DWORDs by 4 DWORDs,
    // tensor_dim0=512, tensor_dim1=16, tile_dim0=512, tile_dim1=16,
    // tensor_dim0_stride=512 elements.
    i32x8 g1;
    g1[0] = (int)((1u << 16)        // data_size = 1 -> 2 bytes
                | (1u << 20)        // pad_enable
                | (7u << 22)        // pad_interval = 7 -> every 256 DWORDs
                | (3u << 25));      // pad_amount   = 3 -> 4 DWORDs (16 B)
    g1[1] = (int)(512u << 16);      // tensor_dim0 low16 (bits 63:48)
    g1[2] = (int)(16u  << 16);      // tensor_dim0 hi=0 | tensor_dim1 low16
    g1[3] = (int)(512u << 16);      // tensor_dim1 hi=0 | tile_dim0=512
    g1[4] = 16;                     // tile_dim1=16, tile_dim2=0
    g1[5] = 512;                    // tensor_dim0_stride low32
    g1[6] = 0;                      // stride hi | tensor_dim1_stride low
    g1[7] = 0;
    i32x4 gz = {0, 0, 0, 0};        // groups 2/3: tile_dim3=0 -> unused (2D)
#if __clang_major__ >= 23
    i32x8 gz8 = {0, 0, 0, 0, 0, 0, 0, 0};
    __builtin_amdgcn_tensor_load_to_lds(g0, g1, gz, gz, gz8, 0);
#else
    __builtin_amdgcn_tensor_load_to_lds(g0, g1, gz, gz, 0);
#endif
    __builtin_amdgcn_s_wait_tensorcnt(0);
  }
  __syncthreads();

  union HU { v16h v; v8h h8[2]; };

  v8f acc[4];
  const v8f vzero = {0.f,0.f,0.f,0.f,0.f,0.f,0.f,0.f};
  #pragma unroll
  for (int i = 0; i < 4; ++i) acc[i] = vzero;

  for (int ks = 0; ks < KST; ++ks) {
    const int k0   = ks * 32;
    // B (KxN) 16-bit layout: lanes 0-15 hold K = k0..k0+7, k0+16..k0+23 of column n=lane,
    // lanes 16-31 hold K = k0+8..k0+15, k0+24..k0+31 of column n=lane-16.
    const int boff = (lane < 16) ? k0 : (k0 + 8);
    HU b;
    b.h8[0] = *(const v8h*)&ldsB[pl * 520 + boff];
    b.h8[1] = *(const v8h*)&ldsB[pl * 520 + boff + 16];
    // Prefetch next k-step of the weight stream (L2-resident, speculative).
    __builtin_prefetch((const void*)(Wp + ((((size_t)wave * KST + ks + 1) * 32 + lane) << 4)), 0, 0);
    #pragma unroll
    for (int mi = 0; mi < 4; ++mi) {
      const int mt = wave + mi * 8;
      const _Float16* ap = Wp + ((((size_t)mt * KST + ks) * 32 + lane) << 4);
      HU a;
      a.h8[0] = *(const v8h*)ap;
      a.h8[1] = *(const v8h*)(ap + 8);
      acc[mi] = __builtin_amdgcn_wmma_f32_16x16x32_f16(
          false, a.v, false, b.v, (short)0, acc[mi], false, false);
    }
  }

  // C/D layout: lane holds column n = lane&15; VGPR r -> M = r (+8 if lane>=16).
  const int n = pbase + pl;
  if (n < SP) {
    const int mro = (lane >= 16) ? 8 : 0;
    #pragma unroll
    for (int mi = 0; mi < 4; ++mi) {
      const int m0 = (wave + mi * 8) * 16 + mro;
      float* op = out + ((size_t)img * SP + n) * NCH + m0;
      const float* rp = resid ? (resid + ((size_t)img * SP + n) * NCH + m0) : nullptr;
      #pragma unroll
      for (int r = 0; r < 8; ++r) {
        float v = acc[mi][r] + bias[m0 + r];
        if (rp) v += rp[r];
        op[r] = v;
      }
    }
  }
}

// ---------------------------------------------------------------------------
// Depthwise 3x3 + bias. Input channel-last f32 [img][196][512];
// output f16 channel-last, zero-padded to SPP pixels.
// ---------------------------------------------------------------------------
__global__ __launch_bounds__(256) void k_dw(const float* __restrict__ Y,
                                            const float* __restrict__ w,
                                            const float* __restrict__ bias,
                                            _Float16* __restrict__ Zt) {
  int img = blockIdx.x, p = blockIdx.y, tid = threadIdx.x;
  for (int ch = tid; ch < NCH; ch += 256) {
    _Float16 o;
    if (p >= SP) {
      o = (_Float16)0.f;
    } else {
      int i = p / HW, j = p % HW;
      float acc = bias[ch];
      #pragma unroll
      for (int r = 0; r < 3; ++r) {
        int ii = i + r - 1;
        if (ii < 0 || ii >= HW) continue;
        #pragma unroll
        for (int s = 0; s < 3; ++s) {
          int jj = j + s - 1;
          if (jj < 0 || jj >= HW) continue;
          acc += w[ch * 9 + r * 3 + s] * Y[((size_t)img * SP + ii * HW + jj) * NCH + ch];
        }
      }
      o = (_Float16)acc;
    }
    Zt[((size_t)img * SPP + p) * NCH + ch] = o;
  }
}

// ---------------------------------------------------------------------------
// Pool(max over pixels) + L2-normalize, for "normal" nodes read from x.
// Output vn1[bt][h][d][c].
// ---------------------------------------------------------------------------
__global__ __launch_bounds__(128) void k_pool_norm_normal_x(const float* __restrict__ x,
                                                            float* __restrict__ vn1) {
  int bid = blockIdx.x;                        // bt*HH*DD
  int bt = bid / (HH * DD); int rem = bid % (HH * DD);
  int h = rem / DD, d = rem % DD;
  int b = bt / TT, t = bt % TT;
  int node = 1 + t * DD + d;
  int c = threadIdx.x;
  const float* src = x + (((size_t)(b * HH + h) * NALL + node) * CC + c) * SP;
  float mx = -3.4e38f;
  for (int p = 0; p < SP; ++p) mx = fmaxf(mx, src[p]);
  __shared__ float red[128];
  red[c] = mx * mx; __syncthreads();
  for (int s = 64; s > 0; s >>= 1) { if (c < s) red[c] += red[c + s]; __syncthreads(); }
  float denom = fmaxf(sqrtf(red[0]), 1e-12f);
  vn1[((size_t)(bt * HH + h) * DD + d) * CC + c] = mx / denom;
}

// Pool+normalize over all 57 nodes (master from x node0 OR channel-last
// master buffer; temp nodes from a channel-last buffer). -> vn[b][h][node][c]
__global__ __launch_bounds__(128) void k_pool_norm_allnode(const float* __restrict__ x,
                                                           const float* __restrict__ master_cl,
                                                           const float* __restrict__ temp,
                                                           float* __restrict__ vn) {
  int bid = blockIdx.x;
  int b = bid / (HH * NALL); int rem = bid % (HH * NALL);
  int h = rem / NALL, node = rem % NALL;
  int c = threadIdx.x;
  float mx = -3.4e38f;
  if (node == 0) {
    if (master_cl) {
      const float* src = master_cl + (size_t)b * SP * NCH + h * CC + c;
      for (int p = 0; p < SP; ++p) mx = fmaxf(mx, src[(size_t)p * NCH]);
    } else {
      const float* src = x + (((size_t)(b * HH + h) * NALL) * CC + c) * SP;
      for (int p = 0; p < SP; ++p) mx = fmaxf(mx, src[p]);
    }
  } else {
    int np = node - 1, t = np / DD, d = np % DD;
    int img = (b * TT + t) * DD + d;
    const float* src = temp + (size_t)img * SP * NCH + h * CC + c;
    for (int p = 0; p < SP; ++p) mx = fmaxf(mx, src[(size_t)p * NCH]);
  }
  __shared__ float red[128];
  red[c] = mx * mx; __syncthreads();
  for (int s = 64; s > 0; s >>= 1) { if (c < s) red[c] += red[c + s]; __syncthreads(); }
  float denom = fmaxf(sqrtf(red[0]), 1e-12f);
  vn[((size_t)(b * HH + h) * NALL + node) * CC + c] = mx / denom;
}

// Generic pool+normalize from a channel-last [img][196][512] buffer.
// mode 0: vo  (bt,h,d)     img=bt*DD+d
// mode 1: mv  (b,h,t)      img=(b*TT+t)*DD+DD-1
// mode 2: vd  (bt,h,m)     m=k*DD+d, s=idx[b,h,t,k], img=(b*TT+s)*DD+d
__global__ __launch_bounds__(128) void k_pool_norm_cl(const float* __restrict__ buf,
                                                      const int* __restrict__ idx,
                                                      float* __restrict__ out,
                                                      int mode) {
  int bid = blockIdx.x, c = threadIdx.x;
  int img = 0, h = 0; size_t oidx = 0;
  if (mode == 0) {
    int bt = bid / (HH * DD); int rem = bid % (HH * DD);
    h = rem / DD; int d = rem % DD;
    img = bt * DD + d;
    oidx = ((size_t)(bt * HH + h) * DD + d) * CC + c;
  } else if (mode == 1) {
    int b = bid / (HH * TT); int rem = bid % (HH * TT);
    h = rem / TT; int t = rem % TT;
    img = (b * TT + t) * DD + (DD - 1);
    oidx = ((size_t)(b * HH + h) * TT + t) * CC + c;
  } else {
    int bt = bid / (HH * KK * DD); int rem = bid % (HH * KK * DD);
    h = rem / (KK * DD); int m = rem % (KK * DD);
    int k = m / DD, d = m % DD;
    int b = bt / TT, t = bt % TT;
    int s = idx[((size_t)(b * HH + h) * TT + t) * KK + k];
    img = (b * TT + s) * DD + d;
    oidx = ((size_t)(bt * HH + h) * (KK * DD) + m) * CC + c;
  }
  const float* src = buf + (size_t)img * SP * NCH + h * CC + c;
  float mx = -3.4e38f;
  for (int p = 0; p < SP; ++p) mx = fmaxf(mx, src[(size_t)p * NCH]);
  __shared__ float red[128];
  red[c] = mx * mx; __syncthreads();
  for (int s = 64; s > 0; s >>= 1) { if (c < s) red[c] += red[c + s]; __syncthreads(); }
  float denom = fmaxf(sqrtf(red[0]), 1e-12f);
  out[oidx] = mx / denom;
}

// A_if[bt,h] = softmax_m( vn.vn / alpha + bias_if ) over 7x7.
__global__ __launch_bounds__(64) void k_att_if(const float* __restrict__ vn1,
                                               const float* __restrict__ alpha,
                                               const float* __restrict__ bias,
                                               float* __restrict__ A) {
  int bid = blockIdx.x; int bt = bid / HH, h = bid % HH;
  int tid = threadIdx.x;
  __shared__ float vns[DD * CC];
  __shared__ float lg[DD * DD];
  const float* base = vn1 + (size_t)(bt * HH + h) * DD * CC;
  for (int i = tid; i < DD * CC; i += 64) vns[i] = base[i];
  __syncthreads();
  float ia = 1.0f / alpha[0];
  if (tid < DD * DD) {
    int n = tid / DD, m = tid % DD;
    float s = 0.f;
    for (int c = 0; c < CC; ++c) s += vns[n * CC + c] * vns[m * CC + c];
    lg[tid] = s * ia + bias[n * DD + m];
  }
  __syncthreads();
  if (tid < DD) {
    int n = tid;
    float mx = -3.4e38f;
    for (int m = 0; m < DD; ++m) mx = fmaxf(mx, lg[n * DD + m]);
    float sum = 0.f, e[DD];
    for (int m = 0; m < DD; ++m) { e[m] = expf(lg[n * DD + m] - mx); sum += e[m]; }
    float* ao = A + (size_t)((bt * HH + h) * DD + n) * DD;
    for (int m = 0; m < DD; ++m) ao[m] = e[m] / sum;
  }
}

// A_tm[b,h,m] = softmax_m( vn0.vnm / alpha + bias[0][m] ) over 57.
__global__ __launch_bounds__(64) void k_att_master(const float* __restrict__ vn,
                                                   const float* __restrict__ alpha,
                                                   const float* __restrict__ bias,
                                                   float* __restrict__ A) {
  int bid = blockIdx.x;            // b*HH+h
  int tid = threadIdx.x;
  __shared__ float v0[CC];
  __shared__ float lg[NALL];
  __shared__ float mx_s, sum_s;
  const float* base = vn + (size_t)bid * NALL * CC;
  for (int i = tid; i < CC; i += 64) v0[i] = base[i];
  __syncthreads();
  float ia = 1.0f / alpha[0];
  if (tid < NALL) {
    const float* vm = base + (size_t)tid * CC;
    float s = 0.f;
    for (int c = 0; c < CC; ++c) s += v0[c] * vm[c];
    lg[tid] = s * ia + bias[tid];
  }
  __syncthreads();
  if (tid == 0) {
    float mx = -3.4e38f;
    for (int m = 0; m < NALL; ++m) mx = fmaxf(mx, lg[m]);
    float sum = 0.f;
    for (int m = 0; m < NALL; ++m) sum += expf(lg[m] - mx);
    mx_s = mx; sum_s = sum;
  }
  __syncthreads();
  if (tid < NALL) A[(size_t)bid * NALL + tid] = expf(lg[tid] - mx_s) / sum_s;
}

// A_rev[bt,h,n,m] = softmax_m( -vo.vd / alpha + bias_rev ) over 7x21.
__global__ __launch_bounds__(256) void k_att_rev(const float* __restrict__ vo,
                                                 const float* __restrict__ vd,
                                                 const float* __restrict__ alpha,
                                                 const float* __restrict__ bias,
                                                 float* __restrict__ A) {
  int bid = blockIdx.x;            // bt*HH+h
  int tid = threadIdx.x;
  __shared__ float vos[DD * CC];
  __shared__ float vds[KK * DD * CC];
  __shared__ float lg[DD * KK * DD];
  const float* ob = vo + (size_t)bid * DD * CC;
  const float* db = vd + (size_t)bid * (KK * DD) * CC;
  for (int i = tid; i < DD * CC; i += 256) vos[i] = ob[i];
  for (int i = tid; i < KK * DD * CC; i += 256) vds[i] = db[i];
  __syncthreads();
  float ia = 1.0f / alpha[0];
  if (tid < DD * KK * DD) {
    int n = tid / (KK * DD), m = tid % (KK * DD);
    float s = 0.f;
    for (int c = 0; c < CC; ++c) s += vos[n * CC + c] * vds[m * CC + c];
    lg[tid] = -s * ia + bias[n * (KK * DD) + m];
  }
  __syncthreads();
  if (tid < DD) {
    int n = tid;
    float mx = -3.4e38f;
    for (int m = 0; m < KK * DD; ++m) mx = fmaxf(mx, lg[n * KK * DD + m]);
    float sum = 0.f;
    for (int m = 0; m < KK * DD; ++m) sum += expf(lg[n * KK * DD + m] - mx);
    float* ao = A + (size_t)(bid * DD + n) * (KK * DD);
    for (int m = 0; m < KK * DD; ++m) ao[m] = expf(lg[n * KK * DD + m] - mx) / sum;
  }
}

// Stable 3-smallest-similarity selection (== top_k(-Asim, 3) indices).
__global__ __launch_bounds__(64) void k_topk(const float* __restrict__ vmain,
                                             int* __restrict__ idx) {
  int bid = blockIdx.x;            // b*HH+h
  int tid = threadIdx.x;
  __shared__ float mv[TT * CC];
  __shared__ float sim[TT * TT];
  const float* base = vmain + (size_t)bid * TT * CC;
  for (int i = tid; i < TT * CC; i += 64) mv[i] = base[i];
  __syncthreads();
  {
    int t = tid / TT, s = tid % TT;
    float v = 0.f;
    for (int c = 0; c < CC; ++c) v += mv[t * CC + c] * mv[s * CC + c];
    sim[tid] = v;
  }
  __syncthreads();
  if (tid < TT) {
    int t = tid;
    bool used[TT];
    for (int s = 0; s < TT; ++s) used[s] = false;
    for (int k = 0; k < KK; ++k) {
      int best = 0; float bv = 3.4e38f;
      for (int s = 0; s < TT; ++s)
        if (!used[s] && sim[t * TT + s] < bv) { bv = sim[t * TT + s]; best = s; }
      used[best] = true;
      idx[((size_t)bid * TT + t) * KK + k] = best;
    }
  }
}

// Node mixing (einsum 'bhnm,bhmcij->bhncij') for the "if" stage; writes
// channel-last Y[img=bt*DD+n][p][ch=h*CC+c].
__global__ __launch_bounds__(256) void k_mix_if(const float* __restrict__ x,
                                                const float* __restrict__ A,
                                                float* __restrict__ Y) {
  int img = blockIdx.x, p = blockIdx.y;
  int bt = img / DD, n = img % DD, b = bt / TT, t = bt % TT;
  int tid = threadIdx.x;
  __shared__ float As[HH * DD];
  if (tid < HH * DD) {
    int h = tid / DD, m = tid % DD;
    As[tid] = A[(size_t)((bt * HH + h) * DD + n) * DD + m];
  }
  __syncthreads();
  for (int ch = tid; ch < NCH; ch += 256) {
    int h = ch >> 7, c = ch & 127;
    float acc = 0.f;
    for (int m = 0; m < DD; ++m) {
      int node = 1 + t * DD + m;
      acc += As[h * DD + m] * x[(((size_t)(b * HH + h) * NALL + node) * CC + c) * SP + p];
    }
    Y[((size_t)img * SP + p) * NCH + ch] = acc;
  }
}

// Mixing over all 57 nodes for the tm1/tm2 master stages (4 output images).
__global__ __launch_bounds__(256) void k_mix_master(const float* __restrict__ x,
                                                    const float* __restrict__ master_cl,
                                                    const float* __restrict__ temp,
                                                    const float* __restrict__ A,
                                                    float* __restrict__ Y) {
  int b = blockIdx.x, p = blockIdx.y, tid = threadIdx.x;
  __shared__ float As[HH * NALL];
  for (int i = tid; i < HH * NALL; i += 256) {
    int h = i / NALL, m = i % NALL;
    As[i] = A[(size_t)(b * HH + h) * NALL + m];
  }
  __syncthreads();
  for (int ch = tid; ch < NCH; ch += 256) {
    int h = ch >> 7, c = ch & 127;
    float m0 = master_cl ? master_cl[((size_t)b * SP + p) * NCH + ch]
                         : x[(((size_t)(b * HH + h) * NALL) * CC + c) * SP + p];
    float acc = As[h * NALL] * m0;
    for (int np = 0; np < TT * DD; ++np) {
      int t = np / DD, d = np % DD;
      int img = (b * TT + t) * DD + d;
      acc += As[h * NALL + 1 + np] * temp[((size_t)img * SP + p) * NCH + ch];
    }
    Y[((size_t)b * SP + p) * NCH + ch] = acc;
  }
}

// Mixing over the gathered "different" set (21 nodes via top-k indices).
__global__ __launch_bounds__(256) void k_mix_rev(const float* __restrict__ N2,
                                                 const int* __restrict__ idx,
                                                 const float* __restrict__ A,
                                                 float* __restrict__ Y) {
  int img = blockIdx.x, p = blockIdx.y;
  int bt = img / DD, n = img % DD, b = bt / TT, t = bt % TT;
  int tid = threadIdx.x;
  __shared__ float As[HH * KK * DD];
  __shared__ int   Sm[HH * KK * DD];
  if (tid < HH * KK * DD) {
    int h = tid / (KK * DD), m = tid % (KK * DD);
    As[tid] = A[((size_t)(bt * HH + h) * DD + n) * (KK * DD) + m];
    int k = m / DD, d = m % DD;
    int s = idx[((size_t)(b * HH + h) * TT + t) * KK + k];
    Sm[tid] = (b * TT + s) * DD + d;
  }
  __syncthreads();
  for (int ch = tid; ch < NCH; ch += 256) {
    int h = ch >> 7;
    float acc = 0.f;
    for (int m = 0; m < KK * DD; ++m)
      acc += As[h * KK * DD + m] * N2[((size_t)Sm[h * KK * DD + m] * SP + p) * NCH + ch];
    Y[((size_t)img * SP + p) * NCH + ch] = acc;
  }
}

// Final transpose: master[b][p][h*C+c] -> out[b][h][c][p].
__global__ __launch_bounds__(256) void k_final(const float* __restrict__ M2,
                                               float* __restrict__ out) {
  int b = blockIdx.x, p = blockIdx.y, tid = threadIdx.x;
  for (int ch = tid; ch < NCH; ch += 256) {
    int h = ch >> 7, c = ch & 127;
    out[(((size_t)(b * HH + h) * CC) + c) * SP + p] = M2[((size_t)b * SP + p) * NCH + ch];
  }
}

} // namespace rcg

extern "C" void kernel_launch(void* const* d_in, const int* in_sizes, int n_in,
                              void* d_out, int out_size, void* d_ws, size_t ws_size,
                              hipStream_t stream) {
  using namespace rcg;
  (void)in_sizes; (void)n_in; (void)out_size; (void)ws_size;

  const float* x      = (const float*)d_in[0];
  const float* a_if   = (const float*)d_in[1];
  const float* a_tm1  = (const float*)d_in[2];
  const float* a_rev  = (const float*)d_in[3];
  const float* a_tm2  = (const float*)d_in[4];
  const float* b_if   = (const float*)d_in[5];
  const float* b_tm1  = (const float*)d_in[6];
  const float* b_rev  = (const float*)d_in[7];
  const float* b_tm2  = (const float*)d_in[8];
  const float* if_dw  = (const float*)d_in[9];
  const float* if_dwb = (const float*)d_in[10];
  const float* if_pw  = (const float*)d_in[11];
  const float* if_pwb = (const float*)d_in[12];
  const float* tm1_dw = (const float*)d_in[13];
  const float* tm1_dwb= (const float*)d_in[14];
  const float* tm1_pw = (const float*)d_in[15];
  const float* tm1_pwb= (const float*)d_in[16];
  const float* rev_dw = (const float*)d_in[17];
  const float* rev_dwb= (const float*)d_in[18];
  const float* rev_pw = (const float*)d_in[19];
  const float* rev_pwb= (const float*)d_in[20];
  const float* tm2_dw = (const float*)d_in[21];
  const float* tm2_dwb= (const float*)d_in[22];
  const float* tm2_pw = (const float*)d_in[23];
  const float* tm2_pwb= (const float*)d_in[24];
  const float* tr_dw  = (const float*)d_in[25];
  const float* tr_dwb = (const float*)d_in[26];
  const float* tr_pw  = (const float*)d_in[27];
  const float* tr_pwb = (const float*)d_in[28];
  float* out = (float*)d_out;

  // -------- workspace carve --------
  char* wp = (char*)d_ws;
  auto alloc = [&](size_t bytes) -> void* {
    void* r = (void*)wp;
    wp += (bytes + 255) & ~(size_t)255;
    return r;
  };
  const size_t big   = (size_t)NIMG * SP * NCH * sizeof(float);   // ~90 MB
  const size_t zbig  = (size_t)NIMG * SPP * NCH * sizeof(_Float16);
  _Float16* WpIF  = (_Float16*)alloc((size_t)NCH * NCH * 2);
  _Float16* WpTM1 = (_Float16*)alloc((size_t)NCH * NCH * 2);
  _Float16* WpREV = (_Float16*)alloc((size_t)NCH * NCH * 2);
  _Float16* WpTM2 = (_Float16*)alloc((size_t)NCH * NCH * 2);
  _Float16* WpTR  = (_Float16*)alloc((size_t)NCH * NCH * 2);
  float* Ybuf   = (float*)alloc(big);                  // mix output (reused)
  _Float16* Zt  = (_Float16*)alloc(zbig);              // depthwise output f16
  float* OUTif  = (float*)alloc(big);                  // if-stage out, reused as NORM3
  float* NORM2  = (float*)alloc(big);                  // tr-stage out
  float* Ymst   = (float*)alloc((size_t)BB * SP * NCH * sizeof(float));
  float* MST1   = (float*)alloc((size_t)BB * SP * NCH * sizeof(float));
  float* MST2   = (float*)alloc((size_t)BB * SP * NCH * sizeof(float));
  float* vn1    = (float*)alloc((size_t)BT * HH * DD * CC * sizeof(float));
  float* Aif    = (float*)alloc((size_t)BT * HH * DD * DD * sizeof(float));
  float* vnall  = (float*)alloc((size_t)BB * HH * NALL * CC * sizeof(float));
  float* Atm    = (float*)alloc((size_t)BB * HH * NALL * sizeof(float));
  float* vmain  = (float*)alloc((size_t)BB * HH * TT * CC * sizeof(float));
  int*   idx    = (int*)alloc((size_t)BB * HH * TT * KK * sizeof(int));
  float* vo     = (float*)alloc((size_t)BT * HH * DD * CC * sizeof(float));
  float* vd     = (float*)alloc((size_t)BT * HH * KK * DD * CC * sizeof(float));
  float* Arev   = (float*)alloc((size_t)BT * HH * DD * KK * DD * sizeof(float));
  float* NORM3  = OUTif;   // OUTif is dead after the tr stage

  // -------- pack all pointwise weights to WMMA A-layout f16 --------
  k_pack_w<<<1024, 256, 0, stream>>>(if_pw,  WpIF);
  k_pack_w<<<1024, 256, 0, stream>>>(tm1_pw, WpTM1);
  k_pack_w<<<1024, 256, 0, stream>>>(rev_pw, WpREV);
  k_pack_w<<<1024, 256, 0, stream>>>(tm2_pw, WpTM2);
  k_pack_w<<<1024, 256, 0, stream>>>(tr_pw,  WpTR);

  // -------- stage 1: "if" attention + sepconv over normal nodes --------
  k_pool_norm_normal_x<<<BT * HH * DD, 128, 0, stream>>>(x, vn1);
  k_att_if<<<BT * HH, 64, 0, stream>>>(vn1, a_if, b_if, Aif);
  k_mix_if<<<dim3(NIMG, SP), 256, 0, stream>>>(x, Aif, Ybuf);
  k_dw<<<dim3(NIMG, SPP), 256, 0, stream>>>(Ybuf, if_dw, if_dwb, Zt);
  k_pw_gemm<<<dim3(NIMG, NTIL), 256, 0, stream>>>(WpIF, Zt, if_pwb, nullptr, OUTif);

  // -------- stage 2: tm1 master update (all_node = [x-master, OUTif]) -----
  k_pool_norm_allnode<<<BB * HH * NALL, 128, 0, stream>>>(x, nullptr, OUTif, vnall);
  k_att_master<<<BB * HH, 64, 0, stream>>>(vnall, a_tm1, b_tm1, Atm);
  k_mix_master<<<dim3(BB, SP), 256, 0, stream>>>(x, nullptr, OUTif, Atm, Ymst);
  k_dw<<<dim3(BB, SPP), 256, 0, stream>>>(Ymst, tm1_dw, tm1_dwb, Zt);
  k_pw_gemm<<<dim3(BB, NTIL), 256, 0, stream>>>(WpTM1, Zt, tm1_pwb, nullptr, MST1);

  // -------- stage 3: tr sepconv (layout of OUTif already matches) --------
  k_dw<<<dim3(NIMG, SPP), 256, 0, stream>>>(OUTif, tr_dw, tr_dwb, Zt);
  k_pw_gemm<<<dim3(NIMG, NTIL), 256, 0, stream>>>(WpTR, Zt, tr_pwb, nullptr, NORM2);

  // -------- stage 4: reverse attention over top-k dissimilar streams -----
  k_pool_norm_cl<<<BT * HH * DD, 128, 0, stream>>>(NORM2, nullptr, vo, 0);
  k_pool_norm_cl<<<BB * HH * TT, 128, 0, stream>>>(NORM2, nullptr, vmain, 1);
  k_topk<<<BB * HH, 64, 0, stream>>>(vmain, idx);
  k_pool_norm_cl<<<BT * HH * KK * DD, 128, 0, stream>>>(NORM2, idx, vd, 2);
  k_att_rev<<<BT * HH, 256, 0, stream>>>(vo, vd, a_rev, b_rev, Arev);
  k_mix_rev<<<dim3(NIMG, SP), 256, 0, stream>>>(NORM2, idx, Arev, Ybuf);
  k_dw<<<dim3(NIMG, SPP), 256, 0, stream>>>(Ybuf, rev_dw, rev_dwb, Zt);
  k_pw_gemm<<<dim3(NIMG, NTIL), 256, 0, stream>>>(WpREV, Zt, rev_pwb, NORM2, NORM3);

  // -------- stage 5: tm2 master update (all_node = [MST1, NORM3]) --------
  k_pool_norm_allnode<<<BB * HH * NALL, 128, 0, stream>>>(x, MST1, NORM3, vnall);
  k_att_master<<<BB * HH, 64, 0, stream>>>(vnall, a_tm2, b_tm2, Atm);
  k_mix_master<<<dim3(BB, SP), 256, 0, stream>>>(x, MST1, NORM3, Atm, Ymst);
  k_dw<<<dim3(BB, SPP), 256, 0, stream>>>(Ymst, tm2_dw, tm2_dwb, Zt);
  k_pw_gemm<<<dim3(BB, NTIL), 256, 0, stream>>>(WpTM2, Zt, tm2_pwb, nullptr, MST2);

  // -------- output --------
  k_final<<<dim3(BB, SP), 256, 0, stream>>>(MST2, out);
}
